// SPLINE_COUPLING_HALF_LAYER_78048145702943
// MI455X (gfx1250) — compile-verified, matrix operand order checked by reading, not run on hardware
//
#include <hip/hip_runtime.h>
#include <math.h>

typedef __attribute__((ext_vector_type(16))) _Float16 v16h;
typedef __attribute__((ext_vector_type(8)))  _Float16 f16x8;
typedef __attribute__((ext_vector_type(8)))  float    v8f;

#define ROWS_TOTAL (256*256)        // 65536 rows of 64 features
#define TM 16                       // rows per workgroup
#define NFLOW 32
#define NVAR 64
#define POUT 800

__device__ __forceinline__ float fast_rcp(float v) {
    return __builtin_amdgcn_rcpf(v);    // v_rcp_f32, ~1 ulp
}

__device__ __forceinline__ float silu_f(float v) {
    return v * fast_rcp(1.0f + __expf(-v));
}

__device__ __forceinline__ float sel8(const float* a, int i) {
    float v = a[0];
#pragma unroll
    for (int b = 1; b < 8; ++b) v = (i == b) ? a[b] : v;
    return v;
}

// Load one WMMA fragment (A-layout or B-layout, identical K mapping):
// base points at (row-of-M-or-N)*stride + kstep*32 + 8*(lane>=16), in f16 units.
__device__ __forceinline__ v16h load_frag(const _Float16* base) {
    f16x8 lo = *(const f16x8*)(base);
    f16x8 hi = *(const f16x8*)(base + 16);
    v16h a;
#pragma unroll
    for (int i = 0; i < 8; ++i) { a[i] = lo[i]; a[i + 8] = hi[i]; }
    return a;
}

// ---------------- Kernel 0: f32 -> f16 transpose (K-major -> N-major) ----
__global__ void transpose_to_f16(const float* __restrict__ src,
                                 _Float16* __restrict__ dst,
                                 int K, int N) {
    int idx = blockIdx.x * 256 + threadIdx.x;
    if (idx >= K * N) return;
    int k = idx / N, n = idx % N;
    dst[n * K + k] = (_Float16)src[idx];
}

// ---------------- Kernel 1: fused MLP (WMMA) + RQS spline ----------------
__global__ __launch_bounds__(256, 1) void fused_spline_kernel(
    const float* __restrict__ x,
    const _Float16* __restrict__ W1t,   // 256 x 32  (N-major)
    const _Float16* __restrict__ W2t,   // 256 x 256 (N-major)
    const _Float16* __restrict__ WOt,   // 800 x 256 (N-major)
    const float* __restrict__ b1,
    const float* __restrict__ b2,
    const float* __restrict__ bO,
    float* __restrict__ out,
    float* __restrict__ partials)       // one per block
{
    // LDS layout (bytes):
    //   H2  f16 [0      , 8192 )  16x256
    //   H1  f16 [8192   , 16384)  16x256   (dead after GEMM2)
    //   XC  f16 [16384  , 17408)  16x32    (dead after GEMM1)
    //   P   f32 [8192   , 59392)  16x800   (overlaps H1+XC only)
    //   LJ  f32 [59392  , 59456)  16
    __shared__ __align__(16) unsigned char smem[59456];
    _Float16* H2 = (_Float16*)(smem);
    _Float16* H1 = (_Float16*)(smem + 8192);
    _Float16* XC = (_Float16*)(smem + 16384);
    float*    P  = (float*)(smem + 8192);
    float*    LJ = (float*)(smem + 59392);

    const int t      = threadIdx.x;
    const int wave   = t >> 5;
    const int lane   = t & 31;
    const int lanelo = lane & 15;
    const int hi16   = lane >> 4;       // 0 or 1
    const int row0   = blockIdx.x * TM;

    // ---- stage xc (16x32) as f16 ----
    for (int e = t; e < TM * NFLOW; e += 256) {
        int r = e >> 5, j = e & 31;
        XC[r * NFLOW + j] = (_Float16)x[(size_t)(row0 + r) * NVAR + NFLOW + j];
    }
    __syncthreads();

    // ---- GEMM1: (16x32) @ (32x256) + b1, silu -> H1 f16 ----
    {
        v16h a = load_frag(XC + lanelo * 32 + 8 * hi16);
        for (int c = wave; c < 16; c += 8) {
            int n0 = c * 16;
            v16h b = load_frag(W1t + (size_t)(n0 + lanelo) * 32 + 8 * hi16);
            v8f acc = {};
            acc = __builtin_amdgcn_wmma_f32_16x16x32_f16(
                false, a, false, b, (short)0, acc, false, false);
            float bias = b1[n0 + lanelo];
#pragma unroll
            for (int r = 0; r < 8; ++r) {
                H1[(r + 8 * hi16) * 256 + n0 + lanelo] =
                    (_Float16)silu_f(acc[r] + bias);
            }
        }
    }
    __syncthreads();

    // ---- GEMM2: (16x256) @ (256x256) + b2, silu -> H2 f16 ----
    {
        v16h aK[8];
#pragma unroll
        for (int k = 0; k < 8; ++k)
            aK[k] = load_frag(H1 + lanelo * 256 + k * 32 + 8 * hi16);
        for (int c = wave; c < 16; c += 8) {
            int n0 = c * 16;
            v8f acc = {};
#pragma unroll
            for (int k = 0; k < 8; ++k) {
                v16h b = load_frag(W2t + (size_t)(n0 + lanelo) * 256 + k * 32 + 8 * hi16);
                acc = __builtin_amdgcn_wmma_f32_16x16x32_f16(
                    false, aK[k], false, b, (short)0, acc, false, false);
            }
            float bias = b2[n0 + lanelo];
#pragma unroll
            for (int r = 0; r < 8; ++r) {
                H2[(r + 8 * hi16) * 256 + n0 + lanelo] =
                    (_Float16)silu_f(acc[r] + bias);
            }
        }
    }
    __syncthreads();

    // ---- GEMM3: (16x256) @ (256x800) + bO -> P f32 (LDS) ----
    {
        v16h aK[8];
#pragma unroll
        for (int k = 0; k < 8; ++k)
            aK[k] = load_frag(H2 + lanelo * 256 + k * 32 + 8 * hi16);
        for (int c = wave; c < 50; c += 8) {
            int n0 = c * 16;
            v8f acc = {};
#pragma unroll
            for (int k = 0; k < 8; ++k) {
                v16h b = load_frag(WOt + (size_t)(n0 + lanelo) * 256 + k * 32 + 8 * hi16);
                acc = __builtin_amdgcn_wmma_f32_16x16x32_f16(
                    false, aK[k], false, b, (short)0, acc, false, false);
            }
            float bias = bO[n0 + lanelo];
#pragma unroll
            for (int r = 0; r < 8; ++r)
                P[(r + 8 * hi16) * POUT + n0 + lanelo] = acc[r] + bias;
        }
    }
    __syncthreads();

    // ---- RQS spline: 512 (row,dim) elems, 2 per thread ----
    float ld_acc[2];
#pragma unroll
    for (int pass = 0; pass < 2; ++pass) {
        int e = t + pass * 256;
        int r = e >> 5, d = e & 31;
        const float* p = P + r * POUT;
        float xv = x[(size_t)(row0 + r) * NVAR + d];

        const float fw = 1.0f - 8.0f * 0.001f;
        // widths / heights softmax
        float uw[8], uh[8];
#pragma unroll
        for (int b = 0; b < 8; ++b) { uw[b] = p[d * 8 + b]; uh[b] = p[256 + d * 8 + b]; }
        float mw = uw[0], mh = uh[0];
#pragma unroll
        for (int b = 1; b < 8; ++b) { mw = fmaxf(mw, uw[b]); mh = fmaxf(mh, uh[b]); }
        float ew[8], eh[8], sw = 0.0f, sh = 0.0f;
#pragma unroll
        for (int b = 0; b < 8; ++b) {
            ew[b] = __expf(uw[b] - mw); sw += ew[b];
            eh[b] = __expf(uh[b] - mh); sh += eh[b];
        }
        float rsw = fast_rcp(sw), rsh = fast_rcp(sh);
        float widths[8], heights[8], cumw[8], cumh[8];
        float cw = -1.0f, ch = -1.0f;
#pragma unroll
        for (int b = 0; b < 8; ++b) {
            widths[b]  = (0.001f + fw * ew[b] * rsw) * 2.0f;
            heights[b] = (0.001f + fw * eh[b] * rsh) * 2.0f;
            cumw[b] = cw; cumh[b] = ch;      // left edge of bin b
            cw += widths[b]; ch += heights[b];
        }
        int idx = 0;
#pragma unroll
        for (int k = 1; k < 8; ++k) idx += (xv >= cumw[k]) ? 1 : 0;

        float cw_k = sel8(cumw, idx), w_k = sel8(widths, idx);
        float ch_k = sel8(cumh, idx), h_k = sel8(heights, idx);
        float us_k  = p[512 + d * 9 + idx];
        float us_k1 = p[512 + d * 9 + idx + 1];
        float d_k  = 0.001f + (50.0f - 0.001f) * fast_rcp(1.0f + __expf(-us_k));
        float d_k1 = 0.001f + (50.0f - 0.001f) * fast_rcp(1.0f + __expf(-us_k1));

        float rw_k  = fast_rcp(w_k);
        float theta = (xv - cw_k) * rw_k;
        theta = fminf(fmaxf(theta, 0.0f), 1.0f);
        float s  = h_k * rw_k;
        float om = 1.0f - theta;
        float tt = theta * om;
        float denom = s + (d_k1 + d_k - 2.0f * s) * tt;
        float y = ch_k + h_k * (s * theta * theta + d_k * tt) * fast_rcp(denom);
        float logdet = 2.0f * __logf(s)
                     + __logf(d_k1 * theta * theta + 2.0f * s * tt + d_k * om * om)
                     - 2.0f * __logf(denom);

        out[(size_t)(row0 + r) * NVAR + d] = y;
        ld_acc[pass] = logdet;
    }

    // deterministic xor-shuffle reduction: wave w owns rows w (pass0) and w+8 (pass1)
#pragma unroll
    for (int off = 16; off > 0; off >>= 1) {
        ld_acc[0] += __shfl_xor(ld_acc[0], off, 32);
        ld_acc[1] += __shfl_xor(ld_acc[1], off, 32);
    }
    if (lane == 0) { LJ[wave] = ld_acc[0]; LJ[wave + 8] = ld_acc[1]; }

    // copy conditioner half through
    for (int e = t; e < TM * NFLOW; e += 256) {
        int r = e >> 5, j = e & 31;
        size_t g = (size_t)(row0 + r) * NVAR + NFLOW + j;
        out[g] = x[g];
    }
    __syncthreads();
    if (t == 0) {
        float sum = 0.0f;
#pragma unroll
        for (int r = 0; r < TM; ++r) sum += LJ[r];
        partials[blockIdx.x] = sum;
    }
}

// ---------------- Kernel 2: deterministic ladJ reduce (16 blocks per m) --
__global__ void ladj_reduce_kernel(const float* __restrict__ partials,
                                   float* __restrict__ ladj) {
    int m = blockIdx.x * 256 + threadIdx.x;
    if (m >= 256) return;
    float sum = 0.0f;
#pragma unroll
    for (int b = 0; b < 16; ++b) sum += partials[m * 16 + b];
    ladj[m] = sum;
}

extern "C" void kernel_launch(void* const* d_in, const int* in_sizes, int n_in,
                              void* d_out, int out_size, void* d_ws, size_t ws_size,
                              hipStream_t stream) {
    const float* x  = (const float*)d_in[0];
    const float* W1 = (const float*)d_in[1];
    const float* b1 = (const float*)d_in[2];
    const float* W2 = (const float*)d_in[3];
    const float* b2 = (const float*)d_in[4];
    const float* WO = (const float*)d_in[5];
    const float* bO = (const float*)d_in[6];

    float* out  = (float*)d_out;
    float* ladj = out + (size_t)ROWS_TOTAL * NVAR;   // 4,194,304 floats of out, then 256 ladJ

    // workspace: f16 transposed weights, then per-block partials
    _Float16* W1t = (_Float16*)d_ws;                 //  8192 f16
    _Float16* W2t = W1t + 32 * 256;                  // 65536 f16
    _Float16* WOt = W2t + 256 * 256;                 // 204800 f16
    float* partials = (float*)(WOt + 256 * 800);     //  4096 f32

    transpose_to_f16<<<(32 * 256 + 255) / 256, 256, 0, stream>>>(W1, W1t, 32, 256);
    transpose_to_f16<<<(256 * 256 + 255) / 256, 256, 0, stream>>>(W2, W2t, 256, 256);
    transpose_to_f16<<<(256 * 800 + 255) / 256, 256, 0, stream>>>(WO, WOt, 256, 800);

    fused_spline_kernel<<<ROWS_TOTAL / TM, 256, 0, stream>>>(
        x, W1t, W2t, WOt, b1, b2, bO, out, partials);

    ladj_reduce_kernel<<<1, 256, 0, stream>>>(partials, ladj);
}
